// BatchTopKSAE_49297634623651
// MI455X (gfx1250) — matrix operand order repaired and checked by reading, not compiled, not run on hardware
//
#include <hip/hip_runtime.h>

typedef __bf16  v16bf __attribute__((ext_vector_type(16)));
typedef float   v8f   __attribute__((ext_vector_type(8)));

#define BM 128
#define BN 128
#define BK 32
#define LDSK 40   // padded LDS row stride in shorts (80B) to spread banks

__device__ __forceinline__ unsigned short f2bf(float f) {
  unsigned u = __builtin_bit_cast(unsigned, f);
  u += 0x7FFFu + ((u >> 16) & 1u);          // round-to-nearest-even
  return (unsigned short)(u >> 16);
}

// CDNA5 async memory->LDS DMA (ASYNCcnt-tracked), bypasses VGPRs entirely.
// vdst is the 32-bit LDS byte address (low 32 bits of a generic LDS pointer).
__device__ __forceinline__ void async_load_b128(unsigned ldsAddr, const void* g) {
  unsigned long long ga = (unsigned long long)(uintptr_t)g;
  asm volatile("global_load_async_to_lds_b128 %0, %1, off"
               :: "v"(ldsAddr), "v"(ga) : "memory");
}
__device__ __forceinline__ void wait_async0() {
#if __has_builtin(__builtin_amdgcn_s_wait_asynccnt)
  __builtin_amdgcn_s_wait_asynccnt(0);
#else
  asm volatile("s_wait_asynccnt 0x0" ::: "memory");
#endif
}

// ---------------- zero scratch ----------------
__global__ void k_zero(unsigned* __restrict__ p, int n) {
  int i = blockIdx.x * blockDim.x + threadIdx.x;
  if (i < n) p[i] = 0u;
}

// ---------------- f32 -> bf16, optional per-column subtract ----------------
__global__ void k_convert(const float* __restrict__ src,
                          const float* __restrict__ colSub, int colMask,
                          unsigned short* __restrict__ dst, long long n) {
  long long i4 = ((long long)blockIdx.x * blockDim.x + threadIdx.x) * 4;
  if (i4 >= n) return;
  float4 v = *(const float4*)(src + i4);
  if (colSub) {
    int c0 = (int)(i4 & (long long)colMask);
    float4 s = *(const float4*)(colSub + c0);
    v.x -= s.x; v.y -= s.y; v.z -= s.z; v.w -= s.w;
  }
  unsigned lo = (unsigned)f2bf(v.x) | ((unsigned)f2bf(v.y) << 16);
  unsigned hi = (unsigned)f2bf(v.z) | ((unsigned)f2bf(v.w) << 16);
  *(uint2*)(dst + i4) = make_uint2(lo, hi);
}

// ---------------- WMMA GEMM + bias + ReLU + coarse histogram ----------------
__global__ __launch_bounds__(256)
void k_sae_gemm(const unsigned short* __restrict__ Xbf,  // [B, D] bf16 (x - b_dec)
                const unsigned short* __restrict__ Wbf,  // [F, D] bf16
                const float* __restrict__ b_enc,         // [F]
                float* __restrict__ out,                 // [B, F]
                unsigned* __restrict__ ghist,
                int D, int F) {
  __shared__ unsigned short As[2][BM * LDSK];
  __shared__ unsigned short Bs[2][BN * LDSK];
  __shared__ unsigned int   lhist[4096];

  const int tid  = threadIdx.x;
  const int lane = tid & 31;
  const int wave = tid >> 5;
  const int wm = wave >> 2;   // 0..1  (64 rows each)
  const int wn = wave & 3;    // 0..3  (32 cols each)

  for (int i = tid; i < 4096; i += 256) lhist[i] = 0u;

  const int mBase = blockIdx.y * BM;
  const int nBase = blockIdx.x * BN;

  const int ldRow  = tid >> 1;         // 0..127
  const int ldColH = (tid & 1) * 16;   // shorts

  const unsigned short* gA = Xbf + (size_t)(mBase + ldRow) * D + ldColH;
  const unsigned short* gB = Wbf + (size_t)(nBase + ldRow) * D + ldColH;

  v8f acc[4][2] = {};

  const int kIters = D / BK;

  // per-thread LDS destinations (byte addresses within LDS)
  const unsigned aDst0 = (unsigned)(uintptr_t)&As[0][ldRow * LDSK + ldColH];
  const unsigned aDst1 = (unsigned)(uintptr_t)&As[1][ldRow * LDSK + ldColH];
  const unsigned bDst0 = (unsigned)(uintptr_t)&Bs[0][ldRow * LDSK + ldColH];
  const unsigned bDst1 = (unsigned)(uintptr_t)&Bs[1][ldRow * LDSK + ldColH];

  // issue async DMA for K-tile 0 into stage 0
  async_load_b128(aDst0,      gA);
  async_load_b128(aDst0 + 16, gA + 8);
  async_load_b128(bDst0,      gB);
  async_load_b128(bDst0 + 16, gB + 8);
  wait_async0();
  __syncthreads();

  const int frRow = lane & 15;
  const int aHalf = (lane < 16) ? 0 : 8;    // K offset (shorts) of first A chunk
  const int bHalf = (lane < 16) ? 0 : 8;    // K offset (shorts) of first B chunk

  union Frag { uint4 u[2]; v16bf v; };

  for (int kt = 0; kt < kIters; ++kt) {
    const int s = kt & 1;
    if (kt + 1 < kIters) {
      // async DMA next K-tile into the other stage; lands while we do WMMAs
      const unsigned short* pA = gA + (size_t)(kt + 1) * BK;
      const unsigned short* pB = gB + (size_t)(kt + 1) * BK;
      const unsigned ad = s ? aDst0 : aDst1;
      const unsigned bd = s ? bDst0 : bDst1;
      async_load_b128(ad,      pA);
      async_load_b128(ad + 16, pA + 8);
      async_load_b128(bd,      pB);
      async_load_b128(bd + 16, pB + 8);
    }

    Frag fa[4], fb[2];
#pragma unroll
    for (int j = 0; j < 2; ++j) {
      const unsigned short* p = &Bs[s][(wn * 32 + j * 16 + frRow) * LDSK + bHalf];
      fb[j].u[0] = *(const uint4*)p;          // K = bHalf .. bHalf+7
      fb[j].u[1] = *(const uint4*)(p + 16);   // K = bHalf+16 .. bHalf+23
    }
#pragma unroll
    for (int i = 0; i < 4; ++i) {
      const unsigned short* p = &As[s][(wm * 64 + i * 16 + frRow) * LDSK + aHalf];
      fa[i].u[0] = *(const uint4*)p;          // K = aHalf .. aHalf+7
      fa[i].u[1] = *(const uint4*)(p + 16);   // K = aHalf+16 .. aHalf+23
    }
#pragma unroll
    for (int i = 0; i < 4; ++i)
#pragma unroll
      for (int j = 0; j < 2; ++j)
        acc[i][j] = __builtin_amdgcn_wmma_f32_16x16x32_bf16(
            false, fa[i].v, false, fb[j].v, (short)0, acc[i][j], false, false);

    wait_async0();      // next-stage DMA complete (had whole compute to land)
    __syncthreads();    // publish stage to all waves in WG
  }

  // epilogue: bias + ReLU + store + coarse histogram (positive-float bit bins)
#pragma unroll
  for (int i = 0; i < 4; ++i) {
    const int m0 = mBase + wm * 64 + i * 16 + ((lane < 16) ? 0 : 8);
#pragma unroll
    for (int j = 0; j < 2; ++j) {
      const int n = nBase + wn * 32 + j * 16 + (lane & 15);
      const float bias = b_enc[n];
      float* po = out + (size_t)m0 * F + n;
      v8f c = acc[i][j];
#pragma unroll
      for (int r = 0; r < 8; ++r) {
        float v = c[r] + bias;
        v = v > 0.f ? v : 0.f;
        *po = v;
        po += F;
        unsigned bin = __builtin_bit_cast(unsigned, v) >> 19;  // 0..4095 for v>=0
        atomicAdd(&lhist[bin], 1u);
      }
    }
  }
  __syncthreads();
  for (int i = tid; i < 4096; i += 256) {
    unsigned cn = lhist[i];
    if (cn) atomicAdd(&ghist[i], cn);
  }
}

// ---------------- find coarse threshold bin ----------------
__global__ void k_select1(const unsigned* __restrict__ hist1,
                          unsigned* __restrict__ ctrl,
                          const int* __restrict__ kPtr, int B) {
  if (threadIdx.x != 0 || blockIdx.x != 0) return;
  const unsigned long long nkeep = (unsigned long long)kPtr[0] * (unsigned long long)B;
  unsigned long long cum = 0, above = 0;
  int bstar = 0;
  for (int b = 4095; b >= 1; --b) {
    unsigned long long c = hist1[b];
    if (cum + c >= nkeep) { bstar = b; above = cum; break; }
    cum += c;
    above = cum;
  }
  ctrl[0] = (unsigned)bstar;
  unsigned long long resid = (nkeep > above) ? (nkeep - above) : 0ull;
  ctrl[1] = (unsigned)resid;
}

// ---------------- refinement histogram within boundary bin ----------------
__global__ void k_hist2(const float* __restrict__ acts, long long n4,
                        const unsigned* __restrict__ ctrl,
                        unsigned* __restrict__ ghist2) {
  __shared__ unsigned lh[4096];
  for (int i = threadIdx.x; i < 4096; i += blockDim.x) lh[i] = 0u;
  __syncthreads();
  const unsigned bstar = ctrl[0];
  const float4* p = (const float4*)acts;
  long long i = (long long)blockIdx.x * blockDim.x + threadIdx.x;
  const long long stride = (long long)gridDim.x * blockDim.x;
  for (; i < n4; i += stride) {
    float4 v = p[i];
    unsigned u0 = __builtin_bit_cast(unsigned, v.x);
    unsigned u1 = __builtin_bit_cast(unsigned, v.y);
    unsigned u2 = __builtin_bit_cast(unsigned, v.z);
    unsigned u3 = __builtin_bit_cast(unsigned, v.w);
    if ((u0 >> 19) == bstar) atomicAdd(&lh[(u0 >> 7) & 4095], 1u);
    if ((u1 >> 19) == bstar) atomicAdd(&lh[(u1 >> 7) & 4095], 1u);
    if ((u2 >> 19) == bstar) atomicAdd(&lh[(u2 >> 7) & 4095], 1u);
    if ((u3 >> 19) == bstar) atomicAdd(&lh[(u3 >> 7) & 4095], 1u);
  }
  __syncthreads();
  for (int t = threadIdx.x; t < 4096; t += blockDim.x) {
    unsigned cn = lh[t];
    if (cn) atomicAdd(&ghist2[t], cn);
  }
}

// ---------------- finalize threshold bit-pattern ----------------
__global__ void k_select2(const unsigned* __restrict__ hist2,
                          unsigned* __restrict__ ctrl) {
  if (threadIdx.x != 0 || blockIdx.x != 0) return;
  const unsigned bstar = ctrl[0];
  const unsigned resid = ctrl[1];
  unsigned thresh;
  if (resid == 0) {
    thresh = (bstar + 1u) << 19;
  } else {
    unsigned long long cum = 0;
    int sstar = 0;
    for (int s2 = 4095; s2 >= 0; --s2) {
      cum += hist2[s2];
      if (cum >= resid) { sstar = s2; break; }
    }
    thresh = (bstar << 19) | ((unsigned)sstar << 7);
  }
  ctrl[2] = thresh;
}

// ---------------- apply threshold (keep iff bits >= thresh) ----------------
__global__ void k_apply(float* __restrict__ acts, long long n4,
                        const unsigned* __restrict__ ctrl) {
  const unsigned thresh = ctrl[2];
  long long i = (long long)blockIdx.x * blockDim.x + threadIdx.x;
  if (i >= n4) return;
  float4* p = (float4*)acts;
  float4 v = p[i];
  if (__builtin_bit_cast(unsigned, v.x) < thresh) v.x = 0.f;
  if (__builtin_bit_cast(unsigned, v.y) < thresh) v.y = 0.f;
  if (__builtin_bit_cast(unsigned, v.z) < thresh) v.z = 0.f;
  if (__builtin_bit_cast(unsigned, v.w) < thresh) v.w = 0.f;
  p[i] = v;
}

extern "C" void kernel_launch(void* const* d_in, const int* in_sizes, int n_in,
                              void* d_out, int out_size, void* d_ws, size_t ws_size,
                              hipStream_t stream) {
  const float* x     = (const float*)d_in[0];
  const float* W_enc = (const float*)d_in[1];
  const float* b_enc = (const float*)d_in[2];
  const float* b_dec = (const float*)d_in[3];
  const int*   kPtr  = (const int*)d_in[4];

  const int D = in_sizes[3];       // activation dim (b_dec length) = 2048
  const int F = in_sizes[2];       // dict size (b_enc length)      = 32768
  const int B = in_sizes[0] / D;   // batch                          = 2048

  unsigned short* Xbf = (unsigned short*)d_ws;            // B*D bf16   (8 MB)
  unsigned short* Wbf = Xbf + (size_t)B * D;              // F*D bf16 (128 MB) -> L2-resident
  unsigned* hist1 = (unsigned*)(Wbf + (size_t)F * D);
  unsigned* hist2 = hist1 + 4096;
  unsigned* ctrl  = hist2 + 4096;

  // zero histograms + control words (deterministic across graph replays)
  k_zero<<<(2 * 4096 + 16 + 255) / 256, 256, 0, stream>>>(hist1, 2 * 4096 + 16);

  const long long nX = (long long)B * D;
  k_convert<<<(unsigned)((nX / 4 + 255) / 256), 256, 0, stream>>>(x, b_dec, D - 1, Xbf, nX);
  const long long nW = (long long)F * D;
  k_convert<<<(unsigned)((nW / 4 + 255) / 256), 256, 0, stream>>>(W_enc, nullptr, 0, Wbf, nW);

  dim3 grid(F / BN, B / BM);
  k_sae_gemm<<<grid, 256, 0, stream>>>(Xbf, Wbf, b_enc, (float*)d_out, hist1, D, F);

  k_select1<<<1, 32, 0, stream>>>(hist1, ctrl, kPtr, B);

  const long long n4 = (long long)out_size / 4;
  k_hist2<<<4096, 256, 0, stream>>>((const float*)d_out, n4, ctrl, hist2);
  k_select2<<<1, 32, 0, stream>>>(hist2, ctrl);
  k_apply<<<(unsigned)((n4 + 255) / 256), 256, 0, stream>>>((float*)d_out, n4, ctrl);
}
